// Mamba_30889404793188
// MI455X (gfx1250) — compile-verified, matrix operand order checked by reading, not run on hardware
//
#include <hip/hip_runtime.h>
#include <hip/hip_bf16.h>

// Mamba block forward for MI455X (gfx1250, wave32, WMMA).
//   1) xr   = x @ Win                      (WMMA f16 gemm, 2048x1024 * 1024x4096)
//   2) xc   = silu(causal_dwconv4(xr[:, :2048]) + conv_b)
//   3) xdbl = xc @ Wx                      (WMMA gemm, N=96)
//   4) delta= softplus(xdbl[:, :64] @ Wdt + bdt)  (fused epilogue)
//   5) scan: h[n] = exp(dt*A)*h + dt*u*B ; y = h.C ; fused +u*D, *silu(res)
//   6) out  = y @ Wout                     (WMMA gemm)

typedef __attribute__((ext_vector_type(16))) _Float16 v16h;
typedef __attribute__((ext_vector_type(8)))  _Float16 v8h;
typedef __attribute__((ext_vector_type(4)))  _Float16 v4h;
typedef __attribute__((ext_vector_type(8)))  float    v8f;

#define BB      2
#define LL      1024
#define DMODEL  1024
#define DIN     2048
#define NST     16
#define DTR     64
#define DCONV   4
#define MROWS   (BB * LL)   // 2048

#define LDSPAD  40          // halves per LDS row: 80B, 16B-aligned, 20-dword stride

__device__ __forceinline__ float silu_f(float x) {
  return x * (1.0f / (1.0f + __expf(-x)));
}
// Branchless softplus: both sides computed, merged with a select (no EXEC branch).
__device__ __forceinline__ float softplus_f(float x) {
  const float sp = __logf(1.0f + __expf(fminf(x, 20.0f)));
  return (x > 20.0f) ? x : sp;
}

// ---------------------------------------------------------------------------
// f16-WMMA GEMM, f32 accumulate: C[M,N] = A[M,K] * B[K,N]
// Block = 256 thr (8 waves); block tile = 64(M) x 128(N); K stepped by 32.
// Wave w owns n-tile w and all four 16-row m-sub-tiles -> 4 WMMAs / K-step.
// act: 0 = store raw, 1 = store softplus(val + bias[n])
// Requires: M % 64 == 0, K % 32 == 0, N % 4 == 0 (tiles guarded at 16).
// ---------------------------------------------------------------------------
__global__ __launch_bounds__(256) void mamba_wmma_gemm(
    const float* __restrict__ A, const float* __restrict__ B,
    float* __restrict__ C, int M, int N, int K,
    int lda, int ldb, int ldc,
    const float* __restrict__ bias, int act)
{
  __shared__ __align__(16) _Float16 As[64][LDSPAD];   // A tile, row-major
  __shared__ __align__(16) _Float16 Bt[128][LDSPAD];  // B tile, TRANSPOSED [n][k]

  const int tid  = threadIdx.x;
  const int lane = tid & 31;
  const int wave = tid >> 5;
  const int m0   = blockIdx.y << 6;        // 64-row tile
  const int n0   = blockIdx.x << 7;        // 128-col group
  const int tn   = n0 + (wave << 4);       // this wave's 16-col tile

  const int g  = lane >> 4;                // half-wave group
  const int mr = lane & 15;

  v8f acc0 = {}, acc1 = {}, acc2 = {}, acc3 = {};

  for (int kk = 0; kk < K; kk += 32) {
    // Prefetch next K tile (global_prefetch_b8)
    if (kk + 32 < K) {
      __builtin_prefetch(A + (size_t)(m0 + (tid >> 2)) * lda + kk + 32 + ((tid & 3) << 3), 0, 1);
      __builtin_prefetch(B + (size_t)(kk + 32 + (tid >> 3)) * ldb + n0 + ((tid & 7) << 4), 0, 1);
    }

    // Stage A tile 64x32: 512 float4 chunks, 2/thread, 128-bit global loads,
    // converted and stored as one aligned 8-byte v4h per chunk.
    #pragma unroll
    for (int ch = tid; ch < 512; ch += 256) {
      const int r = ch >> 3;                 // 8 chunks per 32-wide row
      const int c = (ch & 7) << 2;
      const float4 v = *(const float4*)&A[(size_t)(m0 + r) * lda + kk + c];
      const v4h h = { (_Float16)v.x, (_Float16)v.y, (_Float16)v.z, (_Float16)v.w };
      *(v4h*)&As[r][c] = h;
    }
    // Stage B tile 32x128 transposed: 1024 float4 chunks, 4/thread,
    // 128-bit global loads; transpose scatter into Bt[n][k].
    #pragma unroll
    for (int ch = tid; ch < 1024; ch += 256) {
      const int r = ch >> 5;                 // k row, 32 chunks per 128-wide row
      const int c = (ch & 31) << 2;          // n within tile
      const int n = n0 + c;
      float4 v = make_float4(0.f, 0.f, 0.f, 0.f);
      if (n < N) v = *(const float4*)&B[(size_t)(kk + r) * ldb + n];
      Bt[c + 0][r] = (_Float16)v.x;
      Bt[c + 1][r] = (_Float16)v.y;
      Bt[c + 2][r] = (_Float16)v.z;
      Bt[c + 3][r] = (_Float16)v.w;
    }
    __syncthreads();

    if (tn < N) {   // uniform per wave -> EXEC all-ones for WMMA
      // B fragment: lane holds column n = lane&15; element e holds
      // K = (e>=8 ? 16 : 0) + 8*(lane>>4) + (e&7)  -> two contiguous 16B runs.
      const v8h blo = *(const v8h*)&Bt[(wave << 4) + mr][(g << 3)];
      const v8h bhi = *(const v8h*)&Bt[(wave << 4) + mr][16 + (g << 3)];
      const v16h bf = __builtin_shufflevector(blo, bhi,
          0, 1, 2, 3, 4, 5, 6, 7, 8, 9, 10, 11, 12, 13, 14, 15);

      #pragma unroll
      for (int mt = 0; mt < 4; ++mt) {
        const v8h alo = *(const v8h*)&As[(mt << 4) + mr][(g << 3)];
        const v8h ahi = *(const v8h*)&As[(mt << 4) + mr][16 + (g << 3)];
        const v16h af = __builtin_shufflevector(alo, ahi,
            0, 1, 2, 3, 4, 5, 6, 7, 8, 9, 10, 11, 12, 13, 14, 15);
        v8f* accp = (mt == 0) ? &acc0 : (mt == 1) ? &acc1 : (mt == 2) ? &acc2 : &acc3;
        *accp = __builtin_amdgcn_wmma_f32_16x16x32_f16(
            false, af, false, bf, (short)0, *accp, false, false);
      }
    }
    __syncthreads();
  }

  if (tn < N) {
    const int col = tn + mr;
    #pragma unroll
    for (int mt = 0; mt < 4; ++mt) {
      const v8f acc = (mt == 0) ? acc0 : (mt == 1) ? acc1 : (mt == 2) ? acc2 : acc3;
      #pragma unroll
      for (int v = 0; v < 8; ++v) {
        const int row = m0 + (mt << 4) + v + (g << 3);  // D layout: M = v + 8*(lane>>4)
        float val = acc[v];
        if (act == 1) val = softplus_f(val + bias[col]);
        C[(size_t)row * ldc + col] = val;
      }
    }
  }
}

// ---------------------------------------------------------------------------
// Causal depthwise conv1d (width 4, left-pad 3) + bias + SiLU.
// ---------------------------------------------------------------------------
__global__ __launch_bounds__(256) void mamba_conv_silu(
    const float* __restrict__ xr, const float* __restrict__ conv_w,
    const float* __restrict__ conv_b, float* __restrict__ xc)
{
  const int idx = blockIdx.x * 256 + threadIdx.x;  // over BB*LL*DIN
  const int d  = idx & (DIN - 1);
  const int bl = idx >> 11;            // b*LL + l
  const int l  = bl & (LL - 1);
  const int b  = bl >> 10;

  float s = conv_b[d];
  #pragma unroll
  for (int w = 0; w < DCONV; ++w) {
    const int ls = l - (DCONV - 1) + w;
    if (ls >= 0)
      s += conv_w[w * DIN + d] * xr[(size_t)((b << 10) + ls) * (2 * DIN) + d];
  }
  xc[idx] = silu_f(s);
}

// ---------------------------------------------------------------------------
// Selective scan: one lane per (b, d) channel; h[NST] in VGPRs.
// Reads delta from `dy`, writes gated y in-place. B_t/C_t broadcast per wave.
// ---------------------------------------------------------------------------
__global__ __launch_bounds__(256) void mamba_scan(
    const float* __restrict__ xr,   const float* __restrict__ xc,
    const float* __restrict__ xdbl, const float* __restrict__ A_log,
    const float* __restrict__ Dvec, float* __restrict__ dy)
{
  const int t = blockIdx.x * 256 + threadIdx.x;  // over BB*DIN = 4096
  const int d = t & (DIN - 1);
  const int b = t >> 11;

  float Aneg[NST];
  #pragma unroll
  for (int n = 0; n < NST; ++n) Aneg[n] = -__expf(A_log[d * NST + n]);

  float h[NST];
  #pragma unroll
  for (int n = 0; n < NST; ++n) h[n] = 0.0f;

  const float Dd = Dvec[d];

  for (int l = 0; l < LL; ++l) {
    const int row = (b << 10) + l;
    const float dt = dy[(size_t)row * DIN + d];
    const float u  = xc[(size_t)row * DIN + d];
    const float du = dt * u;
    float y = 0.0f;
    const float* bc = xdbl + (size_t)row * (DTR + 2 * NST);
    #pragma unroll
    for (int n = 0; n < NST; ++n) {
      const float Bn = bc[DTR + n];          // broadcast load
      const float Cn = bc[DTR + NST + n];    // broadcast load
      const float dA = __expf(dt * Aneg[n]);
      h[n] = dA * h[n] + du * Bn;
      y += h[n] * Cn;
    }
    const float res = xr[(size_t)row * (2 * DIN) + DIN + d];
    dy[(size_t)row * DIN + d] = (y + u * Dd) * silu_f(res);
  }
}

// ---------------------------------------------------------------------------
extern "C" void kernel_launch(void* const* d_in, const int* in_sizes, int n_in,
                              void* d_out, int out_size, void* d_ws, size_t ws_size,
                              hipStream_t stream) {
  const float* x      = (const float*)d_in[0];
  const float* Win    = (const float*)d_in[1];
  const float* conv_w = (const float*)d_in[2];
  const float* conv_b = (const float*)d_in[3];
  const float* Wx     = (const float*)d_in[4];
  const float* Wdt    = (const float*)d_in[5];
  const float* bdt    = (const float*)d_in[6];
  const float* A_log  = (const float*)d_in[7];
  const float* Dvec   = (const float*)d_in[8];
  const float* Wout   = (const float*)d_in[9];
  float* out = (float*)d_out;

  // Workspace layout (floats), total ~68 MB:
  float* ws    = (float*)d_ws;
  float* xr    = ws;                                   // MROWS x 4096
  float* xc    = xr   + (size_t)MROWS * (2 * DIN);     // MROWS x 2048
  float* xdbl  = xc   + (size_t)MROWS * DIN;           // MROWS x 96
  float* delta = xdbl + (size_t)MROWS * (DTR + 2*NST); // MROWS x 2048 (becomes y)

  // 1) xr = x @ Win   (M=2048, N=4096, K=1024)
  mamba_wmma_gemm<<<dim3((2 * DIN) / 128, MROWS / 64), 256, 0, stream>>>(
      x, Win, xr, MROWS, 2 * DIN, DMODEL, DMODEL, 2 * DIN, 2 * DIN, nullptr, 0);

  // 2) xc = silu(conv(xr[:, :DIN]) + conv_b)
  mamba_conv_silu<<<(BB * LL * DIN) / 256, 256, 0, stream>>>(xr, conv_w, conv_b, xc);

  // 3) xdbl = xc @ Wx   (N=96: one n-group, waves 6..7 idle-guarded)
  mamba_wmma_gemm<<<dim3(1, MROWS / 64), 256, 0, stream>>>(
      xc, Wx, xdbl, MROWS, DTR + 2 * NST, DIN, DIN, DTR + 2 * NST, DTR + 2 * NST,
      nullptr, 0);

  // 4) delta = softplus(xdbl[:, :64] @ Wdt + bdt)   (K=64, fused epilogue)
  mamba_wmma_gemm<<<dim3(DIN / 128, MROWS / 64), 256, 0, stream>>>(
      xdbl, Wdt, delta, MROWS, DIN, DTR, DTR + 2 * NST, DIN, DIN, bdt, 1);

  // 5) selective scan + skip + gate (y written over delta)
  mamba_scan<<<(BB * DIN) / 256, 256, 0, stream>>>(xr, xc, xdbl, A_log, Dvec, delta);

  // 6) out = y @ Wout
  mamba_wmma_gemm<<<dim3(DMODEL / 128, MROWS / 64), 256, 0, stream>>>(
      delta, Wout, out, MROWS, DMODEL, DIN, DIN, DMODEL, DMODEL, nullptr, 0);
}
